// UNetModel_38173669326885
// MI455X (gfx1250) — matrix operand order repaired and checked by reading, not compiled, not run on hardware
//
#include <hip/hip_runtime.h>
#include <hip/hip_bf16.h>

#define NN   100000
#define EE   500000
#define EP   (EE + NN)      // edges + self loops
#define INC  17
#define HC   256            // H*C
#define MC   64
#define OUTC 17

typedef __attribute__((ext_vector_type(16))) __bf16 bf16x16;
typedef __attribute__((ext_vector_type(8)))  __bf16 bf16x8;
typedef __attribute__((ext_vector_type(8)))  float  f32x8;

// fast hardware approximations (v_rcp_f32 / v_rsq_f32) -- avoids IEEE div expansion
__device__ __forceinline__ float fast_rcp(float v){ return __builtin_amdgcn_rcpf(v); }
__device__ __forceinline__ float fast_rsq(float v){ return __builtin_amdgcn_rsqf(v); }
__device__ __forceinline__ float silu_f(float v){ return v * fast_rcp(1.0f + __expf(-v)); }

// order-preserving float->uint key for atomic max over signed floats
__device__ __forceinline__ unsigned fkey(float v){
  unsigned u = __float_as_uint(v);
  return (u & 0x80000000u) ? ~u : (u | 0x80000000u);
}
__device__ __forceinline__ float unfkey(unsigned k){
  return __uint_as_float((k & 0x80000000u) ? (k ^ 0x80000000u) : ~k);
}

__device__ __forceinline__ f32x8 wmma_bf16(bf16x16 a, bf16x16 b, f32x8 c){
  return __builtin_amdgcn_wmma_f32_16x16x32_bf16(false, a, false, b, (short)0, c, false, false);
}

// A fragment (16x32 bf16) from row-major LDS staging buf[16][ldk].
// Layout per ISA 7.12.2: lane<16 holds K=kb..kb+7 (elems 0..7) and 16+kb.. (elems 8..15)
// with kb = (lane>>4)*8; each half is 8 contiguous bf16 -> one 16B LDS read.
__device__ __forceinline__ bf16x16 load_a_frag(const __bf16* buf, int ldk, int kOff, int lane){
  int m  = lane & 15;
  int kb = (lane >> 4) * 8;
  bf16x8 lo = *(const bf16x8*)(buf + m*ldk + kOff + kb);
  bf16x8 hh = *(const bf16x8*)(buf + m*ldk + kOff + 16 + kb);
  bf16x16 a;
#pragma unroll
  for (int i = 0; i < 8; ++i){ a[i] = lo[i]; a[8+i] = hh[i]; }
  return a;
}

// B fragment (32x16 bf16), pre-swizzled so each lane's 16 values are contiguous (32B).
__device__ __forceinline__ bf16x16 load_b_frag(const __bf16* w, int frag, int lane){
  return *(const bf16x16*)(w + (long long)frag*512 + lane*16);
}

// ---- weight pre-swizzle: W[K][Nout] fp32 row-major -> bf16 B-fragments ----
// frag = kStep*nTiles + nTile; lane = N column (mod 16); elem i: K = kStep*32 + (i<8? 8h+i : 16+8h+(i-8)), h = lane>>4
__global__ void __launch_bounds__(128)
k_swizzle(const float* __restrict__ W, __bf16* __restrict__ dst, int K, int Nout, int nTiles)
{
  int frag = blockIdx.x;
  int s = frag / nTiles, t = frag % nTiles;
  for (int el = threadIdx.x; el < 512; el += 128){
    int lane = el >> 4, i = el & 15;
    int n = (lane & 15) + 16*t;
    int k = s*32 + ((i & 8) << 1) + ((lane >> 4) << 3) + (i & 7);
    float v = (k < K && n < Nout) ? W[k*Nout + n] : 0.0f;
    dst[(long long)frag*512 + lane*16 + i] = (__bf16)v;
  }
}

__global__ void __launch_bounds__(256)
k_init(float* __restrict__ agg, float* __restrict__ denom, unsigned* __restrict__ lmk)
{
  long long total = (long long)NN * HC;
  unsigned ninf = fkey(-__builtin_inff());
  for (long long i = (long long)blockIdx.x*blockDim.x + threadIdx.x;
       i < total; i += (long long)gridDim.x*blockDim.x){
    agg[i] = 0.0f;
    if (i < (long long)NN*4){ denom[i] = 0.0f; lmk[i] = ninf; }
  }
}

// ---- per-node stage 1: time MLP + tf projection, xl/xr transforms ----
__global__ void __launch_bounds__(64)
k_node_stage1(const float* __restrict__ x, const int* __restrict__ tsteps,
              const float* __restrict__ b_t1, const float* __restrict__ b_t2,
              const float* __restrict__ bl, const float* __restrict__ b_tf,
              const __bf16* __restrict__ sw_t1, const __bf16* __restrict__ sw_t2,
              const __bf16* __restrict__ sw_tf, const __bf16* __restrict__ sw_l,
              const __bf16* __restrict__ sw_r,
              float* __restrict__ xl, float* __restrict__ xr, float* __restrict__ tfo)
{
  __shared__ __bf16 sA[2][16][256];
  __shared__ __bf16 sB[2][16][256];
  const int wv   = threadIdx.x >> 5;
  const int lane = threadIdx.x & 31;
  __bf16 (* __restrict__ A)[256] = sA[wv];
  __bf16 (* __restrict__ B)[256] = sB[wv];
  const int base = (blockIdx.x*2 + wv) * 16;
  const int m    = lane & 15;
  const int hi   = lane >> 4;
  const int mrow = hi * 8;
  const int node_m = (base + m < NN) ? base + m : NN - 1;

  // timestep embedding -> A[:,0:64] : cols 0..31 cos, 32..63 sin
  {
    float t = (float)tsteps[node_m];
    const float c = -9.210340371976184f / 32.0f;   // -ln(10000)/half
#pragma unroll
    for (int jj = 0; jj < 16; ++jj){
      int j = hi*16 + jj;
      float arg = t * __expf(c * (float)j);
      A[m][j]      = (__bf16)__cosf(arg);
      A[m][32 + j] = (__bf16)__sinf(arg);
    }
  }
  __syncthreads();

  // emb1 = silu(temb @ W_t1 + b_t1)  [16,256] -> B
#pragma unroll 1
  for (int t = 0; t < 16; ++t){
    f32x8 acc = {0.f,0.f,0.f,0.f,0.f,0.f,0.f,0.f};
#pragma unroll
    for (int s = 0; s < 2; ++s)
      acc = wmma_bf16(load_a_frag(&A[0][0], 256, s*32, lane),
                      load_b_frag(sw_t1, s*16 + t, lane), acc);
    int n = m + 16*t;
    float bias = b_t1[n];
#pragma unroll
    for (int r = 0; r < 8; ++r) B[mrow + r][n] = (__bf16)silu_f(acc[r] + bias);
  }
  __syncthreads();

  // semb = silu(emb1 @ W_t2 + b_t2) [16,256] -> A
#pragma unroll 1
  for (int t = 0; t < 16; ++t){
    f32x8 acc = {0.f,0.f,0.f,0.f,0.f,0.f,0.f,0.f};
#pragma unroll
    for (int s = 0; s < 8; ++s)
      acc = wmma_bf16(load_a_frag(&B[0][0], 256, s*32, lane),
                      load_b_frag(sw_t2, s*16 + t, lane), acc);
    int n = m + 16*t;
    float bias = b_t2[n];
#pragma unroll
    for (int r = 0; r < 8; ++r) A[mrow + r][n] = (__bf16)silu_f(acc[r] + bias);
  }
  __syncthreads();

  // tf = semb @ W_tf + b_tf  [16,64] -> global
#pragma unroll 1
  for (int t = 0; t < 4; ++t){
    f32x8 acc = {0.f,0.f,0.f,0.f,0.f,0.f,0.f,0.f};
#pragma unroll
    for (int s = 0; s < 8; ++s)
      acc = wmma_bf16(load_a_frag(&A[0][0], 256, s*32, lane),
                      load_b_frag(sw_tf, s*4 + t, lane), acc);
    int n = m + 16*t;
    float bias = b_tf[n];
#pragma unroll
    for (int r = 0; r < 8; ++r){
      int nd = base + mrow + r;
      if (nd < NN) tfo[(long long)nd*MC + n] = acc[r] + bias;
    }
  }
  __syncthreads();

  // stage x (K=17 padded to 32) -> A[:,0:32]
  if (hi == 0){
#pragma unroll
    for (int k = 0; k < 16; ++k) A[m][k] = (__bf16)x[(long long)node_m*INC + k];
  } else {
    A[m][16] = (__bf16)x[(long long)node_m*INC + 16];
#pragma unroll
    for (int k = 17; k < 32; ++k) A[m][k] = (__bf16)0.0f;
  }
  __syncthreads();

  // xl = x@Wl + bl ; xr = x@Wr  [16,256] -> global
#pragma unroll 1
  for (int t = 0; t < 16; ++t){
    f32x8 accL = {0.f,0.f,0.f,0.f,0.f,0.f,0.f,0.f};
    f32x8 accR = {0.f,0.f,0.f,0.f,0.f,0.f,0.f,0.f};
    bf16x16 a = load_a_frag(&A[0][0], 256, 0, lane);
    accL = wmma_bf16(a, load_b_frag(sw_l, t, lane), accL);
    accR = wmma_bf16(a, load_b_frag(sw_r, t, lane), accR);
    int n = m + 16*t;
    float bias = bl[n];
#pragma unroll
    for (int r = 0; r < 8; ++r){
      int nd = base + mrow + r;
      if (nd < NN){
        xl[(long long)nd*HC + n] = accL[r] + bias;
        xr[(long long)nd*HC + n] = accR[r];
      }
    }
  }
}

// ---- edge phase: one wave per edge, lane covers 8 of 256 channels ----
__global__ void __launch_bounds__(256)
k_edge_logits(const int* __restrict__ ei, const float* __restrict__ xl,
              const float* __restrict__ xr, const float* __restrict__ att,
              float* __restrict__ logits, unsigned* __restrict__ lmk)
{
  long long gt = (long long)blockIdx.x*blockDim.x + threadIdx.x;
  int e = (int)(gt >> 5);
  if (e >= EP) return;
  int lane = threadIdx.x & 31;
  int s, d;
  if (e < EE){ s = ei[e]; d = ei[EE + e]; } else { s = d = e - EE; }
  int j = lane * 8;
  const float* pa = xl + (long long)s*HC + j;
  const float* pb = xr + (long long)d*HC + j;
  const float* pw = att + j;
  float acc = 0.0f;
#pragma unroll
  for (int i = 0; i < 8; ++i){
    float v = pa[i] + pb[i];
    v = (v > 0.0f) ? v : 0.2f * v;     // leaky_relu(0.2)
    acc += v * pw[i];
  }
  acc += __shfl_xor(acc, 1);
  acc += __shfl_xor(acc, 2);
  acc += __shfl_xor(acc, 4);           // sum over 8 lanes of one head
  if ((lane & 7) == 0){
    int h = lane >> 3;
    logits[(long long)e*4 + h] = acc;
    atomicMax(lmk + (long long)d*4 + h, fkey(acc));
  }
}

__global__ void __launch_bounds__(256)
k_edge_exp(const int* __restrict__ ei, const unsigned* __restrict__ lmk,
           float* __restrict__ exbuf, float* __restrict__ denom)
{
  int t = blockIdx.x*blockDim.x + threadIdx.x;
  if (t >= EP*4) return;
  int e = t >> 2, h = t & 3;
  int d = (e < EE) ? ei[EE + e] : (e - EE);
  float ex = __expf(exbuf[t] - unfkey(lmk[d*4 + h]));
  exbuf[t] = ex;
  atomicAdd(denom + d*4 + h, ex);
}

__global__ void __launch_bounds__(256)
k_edge_agg(const int* __restrict__ ei, const float* __restrict__ xl,
           const float* __restrict__ exbuf, const float* __restrict__ denom,
           float* __restrict__ agg)
{
  long long gt = (long long)blockIdx.x*blockDim.x + threadIdx.x;
  int e = (int)(gt >> 5);
  if (e >= EP) return;
  int lane = threadIdx.x & 31;
  int s, d;
  if (e < EE){ s = ei[e]; d = ei[EE + e]; } else { s = d = e - EE; }
  int h = lane >> 3;
  float alpha = exbuf[(long long)e*4 + h] * fast_rcp(denom[d*4 + h]);
  int j = lane * 8;
  const float* src = xl + (long long)s*HC + j;
  float* dst = agg + (long long)d*HC + j;
#pragma unroll
  for (int i = 0; i < 8; ++i) atomicAdd(dst + i, src[i] * alpha);
}

// ---- final per-node: Wp GEMM + SiLU + tf add + GroupNorm(pairs) + SiLU + W_out GEMM ----
__global__ void __launch_bounds__(64)
k_node_final(const float* __restrict__ agg, const float* __restrict__ gbias,
             const float* __restrict__ bp, const float* __restrict__ tfi,
             const float* __restrict__ gn_g, const float* __restrict__ gn_b,
             const float* __restrict__ b_out,
             const __bf16* __restrict__ sw_p, const __bf16* __restrict__ sw_o,
             float* __restrict__ out)
{
  __shared__ __bf16 sH[2][16][256];
  __shared__ float  sT[2][16][64];
  __shared__ __bf16 sG[2][16][64];
  const int wv   = threadIdx.x >> 5;
  const int lane = threadIdx.x & 31;
  const int base = (blockIdx.x*2 + wv) * 16;
  const int m    = lane & 15;
  const int hi   = lane >> 4;
  const int mrow = hi * 8;
  const int node_m = (base + m < NN) ? base + m : NN - 1;

  // stage h = agg + gat_bias (bf16) and tf rows (f32)
  {
    int c0 = hi * 128;
    for (int c = 0; c < 128; ++c)
      sH[wv][m][c0 + c] = (__bf16)(agg[(long long)node_m*HC + c0 + c] + gbias[c0 + c]);
    int t0 = hi * 32;
    for (int c = 0; c < 32; ++c)
      sT[wv][m][t0 + c] = tfi[(long long)node_m*MC + t0 + c];
  }
  __syncthreads();

  // hmid = silu(h @ Wp + bp) + tf ; GroupNorm(groups of 2 channels) ; silu -> sG (bf16)
#pragma unroll 1
  for (int t = 0; t < 4; ++t){
    f32x8 acc = {0.f,0.f,0.f,0.f,0.f,0.f,0.f,0.f};
#pragma unroll
    for (int s = 0; s < 8; ++s)
      acc = wmma_bf16(load_a_frag(&sH[wv][0][0], 256, s*32, lane),
                      load_b_frag(sw_p, s*4 + t, lane), acc);
    int n = m + 16*t;
    float bias = bp[n], gam = gn_g[n], bet = gn_b[n];
#pragma unroll
    for (int r = 0; r < 8; ++r){
      float v = silu_f(acc[r] + bias) + sT[wv][mrow + r][n];
      float p  = __shfl_xor(v, 1);     // channel pair partner (adjacent N column)
      float mu = 0.5f * (v + p);
      float dl = v - mu;               // group of 2: var = dl*dl
      float hn = dl * fast_rsq(dl*dl + 1e-5f) * gam + bet;
      sG[wv][mrow + r][n] = (__bf16)silu_f(hn);
    }
  }
  __syncthreads();

  // out = hn @ W_out + b_out  (N padded 17 -> 32)
#pragma unroll 1
  for (int t = 0; t < 2; ++t){
    f32x8 acc = {0.f,0.f,0.f,0.f,0.f,0.f,0.f,0.f};
#pragma unroll
    for (int s = 0; s < 2; ++s)
      acc = wmma_bf16(load_a_frag(&sG[wv][0][0], 64, s*32, lane),
                      load_b_frag(sw_o, s*2 + t, lane), acc);
    int n = m + 16*t;
    if (n < OUTC){
      float bias = b_out[n];
#pragma unroll
      for (int r = 0; r < 8; ++r){
        int nd = base + mrow + r;
        if (nd < NN) out[(long long)nd*OUTC + n] = acc[r] + bias;
      }
    }
  }
}

extern "C" void kernel_launch(void* const* d_in, const int* in_sizes, int n_in,
                              void* d_out, int out_size, void* d_ws, size_t ws_size,
                              hipStream_t stream)
{
  (void)in_sizes; (void)n_in; (void)out_size; (void)ws_size;
  const float* x     = (const float*)d_in[0];
  const int*   tst   = (const int*)d_in[1];
  const int*   ei    = (const int*)d_in[2];
  const float* W_t1  = (const float*)d_in[3];
  const float* b_t1  = (const float*)d_in[4];
  const float* W_t2  = (const float*)d_in[5];
  const float* b_t2  = (const float*)d_in[6];
  const float* Wl    = (const float*)d_in[7];
  const float* bl    = (const float*)d_in[8];
  const float* Wr    = (const float*)d_in[9];
  const float* att   = (const float*)d_in[10];
  const float* gbias = (const float*)d_in[11];
  const float* Wp    = (const float*)d_in[12];
  const float* bp    = (const float*)d_in[13];
  const float* W_tf  = (const float*)d_in[14];
  const float* b_tf  = (const float*)d_in[15];
  const float* gn_g  = (const float*)d_in[16];
  const float* gn_b  = (const float*)d_in[17];
  const float* W_out = (const float*)d_in[18];
  const float* b_out = (const float*)d_in[19];
  float* out = (float*)d_out;

  char* p = (char*)d_ws;
  auto carve = [&](size_t bytes)->char*{
    char* r = p; p += (bytes + 4095) & ~(size_t)4095; return r;
  };
  float*    xl    = (float*)   carve((size_t)NN*HC*4);
  float*    xr    = (float*)   carve((size_t)NN*HC*4);
  float*    tfb   = (float*)   carve((size_t)NN*MC*4);
  float*    agg   = (float*)   carve((size_t)NN*HC*4);
  float*    exbuf = (float*)   carve((size_t)EP*4*4);
  float*    denom = (float*)   carve((size_t)NN*4*4);
  unsigned* lmk   = (unsigned*)carve((size_t)NN*4*4);
  __bf16* sw_t1 = (__bf16*)carve((size_t)2*16*1024);
  __bf16* sw_t2 = (__bf16*)carve((size_t)8*16*1024);
  __bf16* sw_tf = (__bf16*)carve((size_t)8*4*1024);
  __bf16* sw_l  = (__bf16*)carve((size_t)1*16*1024);
  __bf16* sw_r  = (__bf16*)carve((size_t)1*16*1024);
  __bf16* sw_p  = (__bf16*)carve((size_t)8*4*1024);
  __bf16* sw_o  = (__bf16*)carve((size_t)2*2*1024);

  // weight pre-swizzle into WMMA B-fragment layout
  k_swizzle<<<2*16, 128, 0, stream>>>(W_t1, sw_t1,  64, 256, 16);
  k_swizzle<<<8*16, 128, 0, stream>>>(W_t2, sw_t2, 256, 256, 16);
  k_swizzle<<<8*4,  128, 0, stream>>>(W_tf, sw_tf, 256,  64,  4);
  k_swizzle<<<1*16, 128, 0, stream>>>(Wl,   sw_l,   17, 256, 16);
  k_swizzle<<<1*16, 128, 0, stream>>>(Wr,   sw_r,   17, 256, 16);
  k_swizzle<<<8*4,  128, 0, stream>>>(Wp,   sw_p,  256,  64,  4);
  k_swizzle<<<2*2,  128, 0, stream>>>(W_out,sw_o,   64,  17,  2);

  k_init<<<2048, 256, 0, stream>>>(agg, denom, lmk);

  int nodeBlocks = (NN + 31) / 32;   // 2 waves/block, 16 nodes/wave
  k_node_stage1<<<nodeBlocks, 64, 0, stream>>>(x, tst, b_t1, b_t2, bl, b_tf,
      sw_t1, sw_t2, sw_tf, sw_l, sw_r, xl, xr, tfb);

  long long edgeThreads = (long long)EP * 32;
  k_edge_logits<<<(int)((edgeThreads + 255)/256), 256, 0, stream>>>(ei, xl, xr, att, exbuf, lmk);
  k_edge_exp<<<(EP*4 + 255)/256, 256, 0, stream>>>(ei, lmk, exbuf, denom);
  k_edge_agg<<<(int)((edgeThreads + 255)/256), 256, 0, stream>>>(ei, xl, exbuf, denom, agg);

  k_node_final<<<nodeBlocks, 64, 0, stream>>>(agg, gbias, bp, tfb, gn_g, gn_b, b_out,
                                              sw_p, sw_o, out);
}
